// Test_30013231464820
// MI455X (gfx1250) — matrix operand order repaired and checked by reading, not compiled, not run on hardware
//
#include <hip/hip_runtime.h>
#include <hip/hip_bf16.h>
#include <math.h>

#define BB   2
#define SS   2048
#define DD   1024
#define HH   16
#define HDIM 64

typedef _Float16 half_t;
typedef __attribute__((ext_vector_type(4)))  half_t v4h;
typedef __attribute__((ext_vector_type(8)))  half_t v8h;
typedef __attribute__((ext_vector_type(16))) half_t v16h;
typedef __attribute__((ext_vector_type(8)))  float  v8f;

// ---- fragment loaders (CDNA5 wave32 WMMA 16x16x32 f16 layouts) -------------
// A (16x32, MxK): lane<16 -> row=lane, halves {k0..k0+7, k0+16..k0+23}
//                 lane>=16 -> row=lane-16, halves {k0+8..k0+15, k0+24..k0+31}
static __device__ __forceinline__ v16h a_frag_f32(const float* rowp, int k0, int lane) {
  const int off = k0 + ((lane & 16) ? 8 : 0);
  const float4* p0 = reinterpret_cast<const float4*>(rowp + off);
  const float4* p1 = reinterpret_cast<const float4*>(rowp + off + 16);
  float4 u0 = p0[0], u1 = p0[1], w0 = p1[0], w1 = p1[1];
  v16h a;
  a[0]=(half_t)u0.x; a[1]=(half_t)u0.y; a[2]=(half_t)u0.z; a[3]=(half_t)u0.w;
  a[4]=(half_t)u1.x; a[5]=(half_t)u1.y; a[6]=(half_t)u1.z; a[7]=(half_t)u1.w;
  a[8]=(half_t)w0.x; a[9]=(half_t)w0.y; a[10]=(half_t)w0.z; a[11]=(half_t)w0.w;
  a[12]=(half_t)w1.x; a[13]=(half_t)w1.y; a[14]=(half_t)w1.z; a[15]=(half_t)w1.w;
  return a;
}

static __device__ __forceinline__ v16h a_frag_f16(const half_t* rowp, int k0, int lane) {
  const int off = k0 + ((lane & 16) ? 8 : 0);
  v8h lo = *reinterpret_cast<const v8h*>(rowp + off);
  v8h hi = *reinterpret_cast<const v8h*>(rowp + off + 16);
  v16h a;
#pragma unroll
  for (int i = 0; i < 8; ++i) { a[i] = lo[i]; a[8 + i] = hi[i]; }
  return a;
}

// B (32x16, KxN) from a row-major f16 matrix where B[k][n] = M[n*ld + k]
// lane<16 -> col=nb+lane, k={k0..k0+7, k0+16..k0+23} ; lane>=16 -> col=nb+lane-16, k={k0+8..k0+15, k0+24..k0+31}
static __device__ __forceinline__ v16h b_frag_rows_f16(const half_t* base, int ld, int nb,
                                                       int k0, int lane) {
  const int n = nb + (lane & 15);
  const int k = k0 + ((lane & 16) ? 8 : 0);
  v8h lo = *reinterpret_cast<const v8h*>(base + (size_t)n * ld + k);
  v8h hi = *reinterpret_cast<const v8h*>(base + (size_t)n * ld + k + 16);
  v16h b;
#pragma unroll
  for (int i = 0; i < 8; ++i) { b[i] = lo[i]; b[8 + i] = hi[i]; }
  return b;
}

// ---- kernel 0: one-shot f32 -> f16 conversion of the 4 weight matrices -----
__global__ __launch_bounds__(256) void cvt_w_kernel(
    const float* __restrict__ w0, const float* __restrict__ w1,
    const float* __restrict__ w2, const float* __restrict__ w3,
    half_t* __restrict__ o0, half_t* __restrict__ o1,
    half_t* __restrict__ o2, half_t* __restrict__ o3) {
  const float* src; half_t* dst;
  switch (blockIdx.y) {
    case 0:  src = w0; dst = o0; break;
    case 1:  src = w1; dst = o1; break;
    case 2:  src = w2; dst = o2; break;
    default: src = w3; dst = o3; break;
  }
  const size_t i = ((size_t)blockIdx.x * 256 + threadIdx.x) * 4;
  float4 u = *reinterpret_cast<const float4*>(src + i);
  v4h h; h[0]=(half_t)u.x; h[1]=(half_t)u.y; h[2]=(half_t)u.z; h[3]=(half_t)u.w;
  *reinterpret_cast<v4h*>(dst + i) = h;
}

// ---- kernel 1: fused Q/K/V projections -> f16 (V stored transposed) --------
// Each wave: 16 rows x 128 cols (8 WMMA accumulators), K-loop over 1024.
__global__ __launch_bounds__(128) void qkv_proj_kernel(
    const float* __restrict__ q, const float* __restrict__ k, const float* __restrict__ v,
    const half_t* __restrict__ whq, const float* __restrict__ bq,
    const half_t* __restrict__ whk, const float* __restrict__ bk,
    const half_t* __restrict__ whv, const float* __restrict__ bv,
    half_t* __restrict__ qh, half_t* __restrict__ kh, half_t* __restrict__ vT) {
  const int lane  = threadIdx.x & 31;
  const int wave  = threadIdx.x >> 5;
  const int mbase = blockIdx.x * 16;
  const int nbase = blockIdx.y * 512 + wave * 128;

  const float* X; const half_t* W; const float* bias; half_t* out;
  bool vtrans = false;
  if (blockIdx.z == 0)      { X = q; W = whq; bias = bq; out = qh; }
  else if (blockIdx.z == 1) { X = k; W = whk; bias = bk; out = kh; }
  else                      { X = v; W = whv; bias = bv; out = vT; vtrans = true; }

  const int mrow = mbase + (lane & 15);
  const float* xrow = X + (size_t)mrow * DD;

  v8f acc[8] = {};
  for (int kk = 0; kk < DD; kk += 32) {
    v16h af = a_frag_f32(xrow, kk, lane);
#pragma unroll
    for (int t = 0; t < 8; ++t) {
      v16h bf = b_frag_rows_f16(W, DD, nbase + t * 16, kk, lane);
      acc[t] = __builtin_amdgcn_wmma_f32_16x16x32_f16(false, af, false, bf,
                                                      (short)0, acc[t], false, false);
    }
  }

  // Store epilogue: uniform branch hoisted out of the element loops.
  const int rbase = (lane & 16) ? 8 : 0;
  const int bidx  = mbase >> 11;            // whole 16-row tile is in one batch
  const int s0    = (mbase & (SS - 1)) + rbase;
  if (vtrans) {
    // out[((b*H+h)*HD + hd)*S + s]: 8 rows are contiguous along S -> one v8h store
#pragma unroll
    for (int t = 0; t < 8; ++t) {
      const int n  = nbase + t * 16 + (lane & 15);
      const float bn = bias[n];
      const int hid = n >> 6, hd = n & 63;
      half_t* p = out + (((size_t)bidx * HH + hid) * HDIM + hd) * SS + s0;
      v8h pk;
#pragma unroll
      for (int r = 0; r < 8; ++r) pk[r] = (half_t)(acc[t][r] + bn);
      *reinterpret_cast<v8h*>(p) = pk;
    }
  } else {
    // out[((b*H+h)*S + s)*HD + hd]: stride HDIM between rows
#pragma unroll
    for (int t = 0; t < 8; ++t) {
      const int n  = nbase + t * 16 + (lane & 15);
      const float bn = bias[n];
      const int hid = n >> 6, hd = n & 63;
      half_t* p = out + (((size_t)bidx * HH + hid) * SS + s0) * HDIM + hd;
#pragma unroll
      for (int r = 0; r < 8; ++r) p[(size_t)r * HDIM] = (half_t)(acc[t][r] + bn);
    }
  }
}

// ---- kernel 2: (QK^T/dk, mask) @ V, streamed in 32-key chunks --------------
__global__ __launch_bounds__(32) void attn_kernel(
    const half_t* __restrict__ qh, const half_t* __restrict__ kh,
    const half_t* __restrict__ vT, const unsigned char* __restrict__ mask,
    half_t* __restrict__ ctxh) {
  __shared__ half_t sS[16 * 40];  // 16x32 f16 score tile, padded stride 40
  const int lane  = threadIdx.x & 31;
  const int qbase = blockIdx.x * 16;
  const int bh    = blockIdx.y;
  const int b     = bh >> 4;
  const int hid   = bh & 15;

  const half_t* Qp = qh + (size_t)bh * SS * HDIM;   // [S, HD]
  const half_t* Kp = kh + (size_t)bh * SS * HDIM;   // [S, HD]
  const half_t* Vp = vT + (size_t)bh * HDIM * SS;   // [HD, S]  (transposed)
  const unsigned char* Mp = mask + (size_t)b * SS * SS;

  const int arow = qbase + (lane & 15);
  const v16h aq0 = a_frag_f16(Qp + (size_t)arow * HDIM, 0, lane);
  const v16h aq1 = a_frag_f16(Qp + (size_t)arow * HDIM, 32, lane);

  v8f vc[4] = {};
  const int rbase = (lane & 16) ? 8 : 0;

  for (int kc = 0; kc < SS; kc += 32) {
    if (kc + 32 < SS) {  // prefetch next key/value chunk (global_prefetch_b8)
      __builtin_prefetch(Kp + (size_t)(kc + 32 + lane) * HDIM, 0, 1);
      __builtin_prefetch(Vp + (size_t)lane * SS + kc + 32, 0, 1);
      __builtin_prefetch(Vp + (size_t)(lane + 32) * SS + kc + 32, 0, 1);
    }
    // score block: Q[16x64] @ K_chunk^T[64x32]
    v8f sacc[2] = {};
#pragma unroll
    for (int ks = 0; ks < 2; ++ks) {
      v16h af = ks ? aq1 : aq0;
#pragma unroll
      for (int t = 0; t < 2; ++t) {
        v16h bf = b_frag_rows_f16(Kp, HDIM, kc + t * 16, ks * 32, lane);
        sacc[t] = __builtin_amdgcn_wmma_f32_16x16x32_f16(false, af, false, bf,
                                                         (short)0, sacc[t], false, false);
      }
    }
    // scale (1/sqrt(64)=0.125), mask, convert f16, re-layout via LDS
#pragma unroll
    for (int t = 0; t < 2; ++t) {
      const int key = kc + t * 16 + (lane & 15);
#pragma unroll
      for (int r = 0; r < 8; ++r) {
        const int qi = qbase + rbase + r;
        float val = sacc[t][r] * 0.125f;
        if (Mp[(size_t)qi * SS + key]) val = -INFINITY;
        sS[(rbase + r) * 40 + t * 16 + (lane & 15)] = (half_t)val;
      }
    }
    __syncthreads();
    // ctx += score[16x32] @ V_chunk[32x64]   (B from transposed V: contiguous)
    v16h as = a_frag_f16(sS + (lane & 15) * 40, 0, lane);
#pragma unroll
    for (int t = 0; t < 4; ++t) {
      v16h bf = b_frag_rows_f16(Vp, SS, t * 16, kc, lane);
      vc[t] = __builtin_amdgcn_wmma_f32_16x16x32_f16(false, as, false, bf,
                                                     (short)0, vc[t], false, false);
    }
    __syncthreads();
  }

  // store ctx in merged [B,S,D] f16 layout (stride DD between rows)
#pragma unroll
  for (int t = 0; t < 4; ++t) {
    const int hd = t * 16 + (lane & 15);
    half_t* p = ctxh + ((size_t)b * SS + qbase + rbase) * DD + hid * HDIM + hd;
#pragma unroll
    for (int r = 0; r < 8; ++r) p[(size_t)r * DD] = (half_t)vc[t][r];
  }
}

// ---- kernel 3: output projection + bias + residual + LayerNorm -------------
__global__ __launch_bounds__(256) void oproj_ln_kernel(
    const half_t* __restrict__ ctxh, const half_t* __restrict__ who,
    const float* __restrict__ bo, const float* __restrict__ qres,
    const float* __restrict__ ln_g, const float* __restrict__ ln_b,
    float* __restrict__ out) {
  __shared__ float xbuf[16 * 1024];  // 64 KB (WGP has 320 KB)
  const int lane  = threadIdx.x & 31;
  const int wave  = threadIdx.x >> 5;
  const int mbase = blockIdx.x * 16;
  const int nbase = wave * 128;
  const int mrow  = mbase + (lane & 15);
  const half_t* arow = ctxh + (size_t)mrow * DD;

  v8f acc[8] = {};
  for (int kk = 0; kk < DD; kk += 32) {
    v16h af = a_frag_f16(arow, kk, lane);
#pragma unroll
    for (int t = 0; t < 8; ++t) {
      v16h bf = b_frag_rows_f16(who, DD, nbase + t * 16, kk, lane);
      acc[t] = __builtin_amdgcn_wmma_f32_16x16x32_f16(false, af, false, bf,
                                                      (short)0, acc[t], false, false);
    }
  }

  const int rbase = (lane & 16) ? 8 : 0;
#pragma unroll
  for (int t = 0; t < 8; ++t) {
    const int n = nbase + t * 16 + (lane & 15);
    const float bn = bo[n];
    const float* qp = qres + (size_t)(mbase + rbase) * DD + n;
    float* xp = xbuf + rbase * 1024 + n;
#pragma unroll
    for (int r = 0; r < 8; ++r)
      xp[r * 1024] = acc[t][r] + bn + qp[(size_t)r * DD];
  }
  __syncthreads();

  // LayerNorm: 16 threads per row, shuffle reduction within 16-lane groups
  const int row = threadIdx.x >> 4;
  const int sub = threadIdx.x & 15;
  float s = 0.f, sq = 0.f;
#pragma unroll 8
  for (int i = 0; i < 64; ++i) {
    float x = xbuf[row * 1024 + i * 16 + sub];
    s += x; sq += x * x;
  }
#pragma unroll
  for (int off = 8; off >= 1; off >>= 1) {
    s  += __shfl_xor(s, off, 16);
    sq += __shfl_xor(sq, off, 16);
  }
  const float mu  = s * (1.0f / 1024.0f);
  const float var = sq * (1.0f / 1024.0f) - mu * mu;
  const float rs  = rsqrtf(var + 1e-5f);
#pragma unroll 8
  for (int i = 0; i < 64; ++i) {
    const int c = i * 16 + sub;
    out[(size_t)(mbase + row) * DD + c] =
        ln_g[c] * (xbuf[row * 1024 + c] - mu) * rs + ln_b[c];
  }
}

// ---- host launcher ----------------------------------------------------------
extern "C" void kernel_launch(void* const* d_in, const int* in_sizes, int n_in,
                              void* d_out, int out_size, void* d_ws, size_t ws_size,
                              hipStream_t stream) {
  const float* q = (const float*)d_in[0];
  const float* k = (const float*)d_in[1];
  const float* v = (const float*)d_in[2];
  const unsigned char* mask = (const unsigned char*)d_in[3];
  // d_in[4] = training flag (identity dropout in eval; unused)
  const float* wq_w = (const float*)d_in[5];
  const float* wq_b = (const float*)d_in[6];
  const float* wk_w = (const float*)d_in[7];
  const float* wk_b = (const float*)d_in[8];
  const float* wv_w = (const float*)d_in[9];
  const float* wv_b = (const float*)d_in[10];
  const float* wo_w = (const float*)d_in[11];
  const float* wo_b = (const float*)d_in[12];
  const float* ln_g = (const float*)d_in[13];
  const float* ln_b = (const float*)d_in[14];
  float* out = (float*)d_out;

  char* ws = (char*)d_ws;
  const size_t wbytes = (size_t)DD * DD * sizeof(half_t);      // 2 MB per weight
  const size_t xbytes = (size_t)BB * SS * DD * sizeof(half_t); // 8 MB per tensor
  half_t* whq  = (half_t*)(ws);
  half_t* whk  = (half_t*)(ws + 1 * wbytes);
  half_t* whv  = (half_t*)(ws + 2 * wbytes);
  half_t* who  = (half_t*)(ws + 3 * wbytes);
  half_t* qh   = (half_t*)(ws + 4 * wbytes);
  half_t* kh   = (half_t*)(ws + 4 * wbytes + 1 * xbytes);
  half_t* vT   = (half_t*)(ws + 4 * wbytes + 2 * xbytes);
  half_t* ctxh = (half_t*)(ws + 4 * wbytes + 3 * xbytes);      // total 40 MB

  dim3 g0((DD * DD) / (256 * 4), 4);
  cvt_w_kernel<<<g0, dim3(256), 0, stream>>>(wq_w, wk_w, wv_w, wo_w, whq, whk, whv, who);

  dim3 g1((BB * SS) / 16, DD / 512, 3);
  qkv_proj_kernel<<<g1, dim3(128), 0, stream>>>(q, k, v, whq, wq_b, whk, wk_b,
                                                whv, wv_b, qh, kh, vT);

  dim3 g2(SS / 16, BB * HH);
  attn_kernel<<<g2, dim3(32), 0, stream>>>(qh, kh, vT, mask, ctxh);

  dim3 g3((BB * SS) / 16);
  oproj_ln_kernel<<<g3, dim3(256), 0, stream>>>(ctxh, who, wo_b, q, ln_g, ln_b, out);
}
